// AttentionLayer_70592082477718
// MI455X (gfx1250) — matrix operand order repaired and checked by reading, not compiled, not run on hardware
//
#include <hip/hip_runtime.h>

// ---------------------------------------------------------------------------
// Fused attention layer for MI455X (gfx1250, wave32, WMMA).
//   B=2, S=2048, D=1024, H=16, DK=DV=64
// All GEMMs: v_wmma_f32_16x16x32_bf16, f32 accumulate.
// Strategy: pre-convert every GEMM operand to bf16 in the layout WMMA wants
// (A row-major, B K-major), so GEMM inner loops are pure b128 loads + WMMA.
// d_out = [ y : B*S*D fp32 ][ attn : B*H*S*S fp32 ]
// ---------------------------------------------------------------------------

#define NB    2
#define S_LEN 2048
#define D_DIM 1024
#define NH    16
#define HD    64

typedef __bf16 v8bf  __attribute__((ext_vector_type(8)));
typedef __bf16 v16bf __attribute__((ext_vector_type(16)));
typedef float  v8f   __attribute__((ext_vector_type(8)));

static __device__ __forceinline__ v16bf cat8(v8bf lo, v8bf hi) {
  return __builtin_shufflevector(lo, hi, 0,1,2,3,4,5,6,7,8,9,10,11,12,13,14,15);
}

// 8 fp32 -> 8 bf16 (native v_cvt path)
static __device__ __forceinline__ v8bf cvt8(const float* __restrict__ p) {
  float4 a = *(const float4*)p;
  float4 b = *(const float4*)(p + 4);
  v8bf r;
  r[0]=(__bf16)a.x; r[1]=(__bf16)a.y; r[2]=(__bf16)a.z; r[3]=(__bf16)a.w;
  r[4]=(__bf16)b.x; r[5]=(__bf16)b.y; r[6]=(__bf16)b.z; r[7]=(__bf16)b.w;
  return r;
}

// A fragment (16x32 bf16, MxK) from bf16 row-major [row*ld + k].
// ISA layout: lane L -> row L%16; K halves {h*8..h*8+7} and {16+h*8..}, h=L/16.
static __device__ __forceinline__ v16bf a_frag_bf16(const __bf16* __restrict__ base, int ld) {
  const int lane = threadIdx.x & 31;
  const __bf16* p = base + (lane & 15) * ld + ((lane >> 4) << 3);
  return cat8(*(const v8bf*)p, *(const v8bf*)(p + 16));
}

// A fragment converted on the fly from fp32 row-major (used for attn @ V).
static __device__ __forceinline__ v16bf a_frag_f32(const float* __restrict__ base, int ld) {
  const int lane = threadIdx.x & 31;
  const float* p = base + (lane & 15) * ld + ((lane >> 4) << 3);
  return cat8(cvt8(p), cvt8(p + 16));
}

// B fragment (32x16 bf16, KxN) from K-major storage: element (k,n) at base[n*ld + k].
// ISA layout: lane L -> col L%16; K halves {h*8..h*8+7} and {16+h*8..}, h=L/16.
static __device__ __forceinline__ v16bf b_frag_kmajor(const __bf16* __restrict__ base, int ld) {
  const int lane = threadIdx.x & 31;
  const __bf16* p = base + (lane & 15) * ld + ((lane >> 4) << 3);
  return cat8(*(const v8bf*)p, *(const v8bf*)(p + 16));
}

static __device__ __forceinline__ v8f wmma_bf16(v16bf a, v16bf b, v8f c) {
  return __builtin_amdgcn_wmma_f32_16x16x32_bf16(false, a, false, b, (short)0, c, false, false);
}

// ---------------------------------------------------------------------------
// 0a) Elementwise fp32 -> bf16 (for ip). 8 elems/thread.
// ---------------------------------------------------------------------------
__global__ __launch_bounds__(256)
void cvt_kernel(const float* __restrict__ in, __bf16* __restrict__ out) {
  const size_t i = ((size_t)blockIdx.x * 256 + threadIdx.x) * 8;
  *(v8bf*)(out + i) = cvt8(in + i);
}

// ---------------------------------------------------------------------------
// 0b) Transpose-convert weight [K=1024][N=1024] fp32 -> Wt [N][K] bf16.
//     64x64 tile via LDS; both global sides coalesced.
// ---------------------------------------------------------------------------
__global__ __launch_bounds__(256)
void wtrans_kernel(const float* __restrict__ W, __bf16* __restrict__ Wt) {
  __shared__ __bf16 t[64][65];
  const int tid = threadIdx.x;
  const int n0 = blockIdx.x * 64, k0 = blockIdx.y * 64;
#pragma unroll
  for (int j = 0; j < 16; ++j) {
    int idx = tid + j * 256;
    int kk = idx >> 6, nn = idx & 63;
    t[kk][nn] = (__bf16)W[(size_t)(k0 + kk) * D_DIM + n0 + nn];
  }
  __syncthreads();
#pragma unroll
  for (int j = 0; j < 16; ++j) {
    int idx = tid + j * 256;
    int nn = idx >> 6, kk = idx & 63;
    Wt[(size_t)(n0 + nn) * D_DIM + k0 + kk] = t[kk][nn];
  }
}

// ---------------------------------------------------------------------------
// 0c) Per-head transpose of V: vb[b*S+s][h*64+d] -> vt[(bh*64+d)][s] (bf16).
// ---------------------------------------------------------------------------
__global__ __launch_bounds__(256)
void vtrans_kernel(const __bf16* __restrict__ vb, __bf16* __restrict__ vt) {
  __shared__ __bf16 t[64][65];
  const int tid = threadIdx.x;
  const int s0 = blockIdx.x * 64;
  const int bh = blockIdx.y, b = bh >> 4, h = bh & 15;
#pragma unroll
  for (int j = 0; j < 16; ++j) {
    int idx = tid + j * 256;
    int ss = idx >> 6, dd = idx & 63;
    t[ss][dd] = vb[(size_t)(b * S_LEN + s0 + ss) * D_DIM + h * HD + dd];
  }
  __syncthreads();
#pragma unroll
  for (int j = 0; j < 16; ++j) {
    int idx = tid + j * 256;
    int dd = idx >> 6, ss = idx & 63;
    vt[((size_t)bh * HD + dd) * S_LEN + s0 + ss] = t[ss][dd];
  }
}

// ---------------------------------------------------------------------------
// 1) Projection GEMM: out_bf16[4096,1024] = A_bf16[4096,1024] @ Wt_bf16[N][K]^T
//    8 waves/block; block tile 256x64; per-wave 32x64 (2 A-frags x 4 B-frags).
// ---------------------------------------------------------------------------
__global__ __launch_bounds__(256)
void proj_gemm_kernel(const __bf16* __restrict__ A, const __bf16* __restrict__ Wt,
                      __bf16* __restrict__ out) {
  const int tid = threadIdx.x, lane = tid & 31, wave = tid >> 5;
  const int m0 = blockIdx.x * 256 + wave * 32;
  const int n0 = blockIdx.y * 64;
  v8f acc[2][4] = {};

  for (int k0 = 0; k0 < D_DIM; k0 += 32) {
    v16bf a0 = a_frag_bf16(A + (size_t)m0 * D_DIM + k0, D_DIM);
    v16bf a1 = a_frag_bf16(A + (size_t)(m0 + 16) * D_DIM + k0, D_DIM);
#pragma unroll
    for (int ns = 0; ns < 4; ++ns) {
      v16bf b = b_frag_kmajor(Wt + (size_t)(n0 + ns * 16) * D_DIM + k0, D_DIM);
      acc[0][ns] = wmma_bf16(a0, b, acc[0][ns]);
      acc[1][ns] = wmma_bf16(a1, b, acc[1][ns]);
    }
  }

  const int hi = lane >> 4, nn = lane & 15;
#pragma unroll
  for (int mf = 0; mf < 2; ++mf)
#pragma unroll
    for (int ns = 0; ns < 4; ++ns)
#pragma unroll
      for (int r = 0; r < 8; ++r)
        out[(size_t)(m0 + mf * 16 + hi * 8 + r) * D_DIM + n0 + ns * 16 + nn] =
            (__bf16)acc[mf][ns][r];
}

// ---------------------------------------------------------------------------
// 2) Scores: attn_raw[bh][q][k] = (Q[q,:] . K[k,:]) / 8
//    K rows are already K-major for the B fragment.
// ---------------------------------------------------------------------------
__global__ __launch_bounds__(256)
void scores_kernel(const __bf16* __restrict__ qb, const __bf16* __restrict__ kb,
                   float* __restrict__ attn) {
  const int tid = threadIdx.x, lane = tid & 31, wave = tid >> 5;
  const int q0 = blockIdx.x * 256 + wave * 32;
  const int n0 = blockIdx.y * 64;
  const int bh = blockIdx.z, b = bh >> 4, h = bh & 15;
  const __bf16* Qp = qb + (size_t)(b * S_LEN) * D_DIM + h * HD;
  const __bf16* Kp = kb + (size_t)(b * S_LEN) * D_DIM + h * HD;
  v8f acc[2][4] = {};

#pragma unroll
  for (int kd = 0; kd < HD; kd += 32) {
    v16bf a0 = a_frag_bf16(Qp + (size_t)q0 * D_DIM + kd, D_DIM);
    v16bf a1 = a_frag_bf16(Qp + (size_t)(q0 + 16) * D_DIM + kd, D_DIM);
#pragma unroll
    for (int ns = 0; ns < 4; ++ns) {
      v16bf bf = b_frag_kmajor(Kp + (size_t)(n0 + ns * 16) * D_DIM + kd, D_DIM);
      acc[0][ns] = wmma_bf16(a0, bf, acc[0][ns]);
      acc[1][ns] = wmma_bf16(a1, bf, acc[1][ns]);
    }
  }

  const float sc = 0.125f;                          // 1/sqrt(DK)
  const int hi = lane >> 4, nn = lane & 15;
  float* base = attn + (size_t)bh * S_LEN * S_LEN;
#pragma unroll
  for (int mf = 0; mf < 2; ++mf)
#pragma unroll
    for (int ns = 0; ns < 4; ++ns)
#pragma unroll
      for (int r = 0; r < 8; ++r)
        base[(size_t)(q0 + mf * 16 + hi * 8 + r) * S_LEN + n0 + ns * 16 + nn] =
            acc[mf][ns][r] * sc;
}

// ---------------------------------------------------------------------------
// 3) Row softmax in place: one 256-thread block per row of 2048.
// ---------------------------------------------------------------------------
__global__ __launch_bounds__(256)
void softmax_kernel(float* __restrict__ attn) {
  __shared__ float red[256];
  const int tid = threadIdx.x;
  float* row = attn + (size_t)blockIdx.x * S_LEN;
  float x[8];
  float m = -3.402823466e38f;
#pragma unroll
  for (int j = 0; j < 8; ++j) { x[j] = row[tid + j * 256]; m = fmaxf(m, x[j]); }
  red[tid] = m; __syncthreads();
  for (int s = 128; s > 0; s >>= 1) {
    if (tid < s) red[tid] = fmaxf(red[tid], red[tid + s]);
    __syncthreads();
  }
  m = red[0]; __syncthreads();
  float sum = 0.f;
#pragma unroll
  for (int j = 0; j < 8; ++j) { x[j] = __expf(x[j] - m); sum += x[j]; }
  red[tid] = sum; __syncthreads();
  for (int s = 128; s > 0; s >>= 1) {
    if (tid < s) red[tid] += red[tid + s];
    __syncthreads();
  }
  const float inv = 1.0f / red[0];
#pragma unroll
  for (int j = 0; j < 8; ++j) row[tid + j * 256] = x[j] * inv;
}

// ---------------------------------------------------------------------------
// 4) ctx = attn @ V ; A converted fp32->bf16 in-register (streaming, one use),
//    B from pre-transposed vt (contiguous K). ctx written bf16 [b*S+s][h*64+d].
// ---------------------------------------------------------------------------
__global__ __launch_bounds__(256)
void pv_kernel(const float* __restrict__ attn, const __bf16* __restrict__ vt,
               __bf16* __restrict__ ctx) {
  const int tid = threadIdx.x, lane = tid & 31, wave = tid >> 5;
  const int q0 = blockIdx.x * 256 + wave * 32;
  const int bh = blockIdx.y, b = bh >> 4, h = bh & 15;
  const float* Ap = attn + (size_t)bh * S_LEN * S_LEN;
  const __bf16* Vtp = vt + (size_t)bh * HD * S_LEN;
  v8f acc[2][4] = {};

  for (int k0 = 0; k0 < S_LEN; k0 += 32) {
    // prefetch the streaming attn rows a few k-steps ahead
    __builtin_prefetch(Ap + (size_t)(q0 + (lane & 15)) * S_LEN + k0 + 256, 0, 1);
    v16bf a0 = a_frag_f32(Ap + (size_t)q0 * S_LEN + k0, S_LEN);
    v16bf a1 = a_frag_f32(Ap + (size_t)(q0 + 16) * S_LEN + k0, S_LEN);
#pragma unroll
    for (int ns = 0; ns < 4; ++ns) {
      v16bf bf = b_frag_kmajor(Vtp + (size_t)(ns * 16) * S_LEN + k0, S_LEN);
      acc[0][ns] = wmma_bf16(a0, bf, acc[0][ns]);
      acc[1][ns] = wmma_bf16(a1, bf, acc[1][ns]);
    }
  }

  const int hi = lane >> 4, nn = lane & 15;
#pragma unroll
  for (int mf = 0; mf < 2; ++mf)
#pragma unroll
    for (int ns = 0; ns < 4; ++ns)
#pragma unroll
      for (int r = 0; r < 8; ++r) {
        int q = q0 + mf * 16 + hi * 8 + r;
        ctx[(size_t)(b * S_LEN + q) * D_DIM + h * HD + ns * 16 + nn] =
            (__bf16)acc[mf][ns][r];
      }
}

// ---------------------------------------------------------------------------
// 5) fc: tmp_f32 = ctx_bf16 @ fcw + fc_b + ip   (residual fused)
// ---------------------------------------------------------------------------
__global__ __launch_bounds__(256)
void fc_kernel(const __bf16* __restrict__ ctx, const __bf16* __restrict__ Wt,
               const float* __restrict__ bias, const float* __restrict__ ip,
               float* __restrict__ outb) {
  const int tid = threadIdx.x, lane = tid & 31, wave = tid >> 5;
  const int m0 = blockIdx.x * 256 + wave * 32;
  const int n0 = blockIdx.y * 64;
  v8f acc[2][4] = {};

  for (int k0 = 0; k0 < D_DIM; k0 += 32) {
    v16bf a0 = a_frag_bf16(ctx + (size_t)m0 * D_DIM + k0, D_DIM);
    v16bf a1 = a_frag_bf16(ctx + (size_t)(m0 + 16) * D_DIM + k0, D_DIM);
#pragma unroll
    for (int ns = 0; ns < 4; ++ns) {
      v16bf b = b_frag_kmajor(Wt + (size_t)(n0 + ns * 16) * D_DIM + k0, D_DIM);
      acc[0][ns] = wmma_bf16(a0, b, acc[0][ns]);
      acc[1][ns] = wmma_bf16(a1, b, acc[1][ns]);
    }
  }

  const int hi = lane >> 4, nn = lane & 15;
#pragma unroll
  for (int mf = 0; mf < 2; ++mf)
#pragma unroll
    for (int ns = 0; ns < 4; ++ns)
#pragma unroll
      for (int r = 0; r < 8; ++r) {
        int m = m0 + mf * 16 + hi * 8 + r;
        int n = n0 + ns * 16 + nn;
        outb[(size_t)m * D_DIM + n] = acc[mf][ns][r] + bias[n] + ip[(size_t)m * D_DIM + n];
      }
}

// ---------------------------------------------------------------------------
// 6) LayerNorm over D=1024: one block per row, 4 elems/thread.
// ---------------------------------------------------------------------------
__global__ __launch_bounds__(256)
void ln_kernel(const float* __restrict__ t, const float* __restrict__ g,
               const float* __restrict__ be, float* __restrict__ y) {
  __shared__ float r1[256];
  __shared__ float r2[256];
  const int tid = threadIdx.x;
  const float* x = t + (size_t)blockIdx.x * D_DIM;
  float4 v = *(const float4*)(x + tid * 4);
  float s  = v.x + v.y + v.z + v.w;
  float ss = v.x * v.x + v.y * v.y + v.z * v.z + v.w * v.w;
  r1[tid] = s; r2[tid] = ss; __syncthreads();
  for (int st = 128; st > 0; st >>= 1) {
    if (tid < st) { r1[tid] += r1[tid + st]; r2[tid] += r2[tid + st]; }
    __syncthreads();
  }
  const float mu  = r1[0] * (1.0f / D_DIM);
  const float var = r2[0] * (1.0f / D_DIM) - mu * mu;
  const float rs  = __frsqrt_rn(var + 1e-6f);
  float* out = y + (size_t)blockIdx.x * D_DIM + tid * 4;
  const float* gp = g + tid * 4;
  const float* bp = be + tid * 4;
  out[0] = (v.x - mu) * rs * gp[0] + bp[0];
  out[1] = (v.y - mu) * rs * gp[1] + bp[1];
  out[2] = (v.z - mu) * rs * gp[2] + bp[2];
  out[3] = (v.w - mu) * rs * gp[3] + bp[3];
}

// ---------------------------------------------------------------------------
extern "C" void kernel_launch(void* const* d_in, const int* in_sizes, int n_in,
                              void* d_out, int out_size, void* d_ws, size_t ws_size,
                              hipStream_t stream) {
  const float* ip  = (const float*)d_in[0];
  const float* wk  = (const float*)d_in[1];
  const float* wq  = (const float*)d_in[2];
  const float* wv  = (const float*)d_in[3];
  const float* fcw = (const float*)d_in[4];
  const float* fcb = (const float*)d_in[5];
  const float* lng = (const float*)d_in[6];
  const float* lnb = (const float*)d_in[7];

  const size_t MD = (size_t)NB * S_LEN * D_DIM;     // 4,194,304
  const size_t WD = (size_t)D_DIM * D_DIM;          // 1,048,576
  float* y    = (float*)d_out;
  float* attn = y + MD;

  __bf16* ipb  = (__bf16*)d_ws;
  __bf16* qb   = ipb + MD;
  __bf16* kb   = qb + MD;
  __bf16* vb   = kb + MD;
  __bf16* cx   = vb + MD;
  __bf16* wqt  = cx + MD;
  __bf16* wkt  = wqt + WD;
  __bf16* wvt  = wkt + WD;
  __bf16* fcwt = wvt + WD;
  __bf16* vt   = fcwt + WD;
  float*  tmp  = (float*)(vt + MD);                 // total ws use ~72 MB

  dim3 blk(256);

  cvt_kernel<<<dim3(MD / (256 * 8)), blk, 0, stream>>>(ip, ipb);
  wtrans_kernel<<<dim3(16, 16), blk, 0, stream>>>(wq, wqt);
  wtrans_kernel<<<dim3(16, 16), blk, 0, stream>>>(wk, wkt);
  wtrans_kernel<<<dim3(16, 16), blk, 0, stream>>>(wv, wvt);
  wtrans_kernel<<<dim3(16, 16), blk, 0, stream>>>(fcw, fcwt);

  dim3 gGemm(16, 16);                               // 4096/256 x 1024/64
  proj_gemm_kernel<<<gGemm, blk, 0, stream>>>(ipb, wqt, qb);
  proj_gemm_kernel<<<gGemm, blk, 0, stream>>>(ipb, wkt, kb);
  proj_gemm_kernel<<<gGemm, blk, 0, stream>>>(ipb, wvt, vb);

  vtrans_kernel<<<dim3(S_LEN / 64, NB * NH), blk, 0, stream>>>(vb, vt);

  scores_kernel<<<dim3(S_LEN / 256, S_LEN / 64, NB * NH), blk, 0, stream>>>(qb, kb, attn);
  softmax_kernel<<<dim3(NB * NH * S_LEN), blk, 0, stream>>>(attn);
  pv_kernel<<<dim3(S_LEN / 256, NB * NH), blk, 0, stream>>>(attn, vt, cx);

  fc_kernel<<<gGemm, blk, 0, stream>>>(cx, fcwt, fcb, ip, tmp);
  ln_kernel<<<dim3(NB * S_LEN), blk, 0, stream>>>(tmp, lng, lnb, y);
}
